// FlowMLP_66219805770045
// MI455X (gfx1250) — matrix operand order repaired and checked by reading, compiled, not run on hardware
//
#include <hip/hip_runtime.h>

typedef __attribute__((ext_vector_type(16))) _Float16 v16h;
typedef __attribute__((ext_vector_type(8)))  _Float16 v8h;
typedef __attribute__((ext_vector_type(8)))  float    v8f;

namespace {
constexpr int B_   = 4;
constexpr int C_   = 192;
constexpr int H_   = 48;
constexpr int W_   = 48;
constexpr int DN_  = 384;       // d_inner
constexpr int NS_  = 16;        // ssm state
constexpr int R_   = 12;        // dt_rank
constexpr int K_   = 4;         // scan directions
constexpr int MH_  = 768;       // mlp hidden
constexpr int L_   = H_ * W_;   // 2304
constexpr int MTOK = B_ * L_;   // 9216
constexpr int NPAD = 48;        // R + 2N = 44 padded to 48
}

static __device__ __forceinline__ float gelu_exact(float x) {
    return 0.5f * x * (1.0f + erff(x * 0.70710678118654752f));
}
static __device__ __forceinline__ float silu_f(float x) {
    return x / (1.0f + __expf(-x));
}

// ---------------- weight conversion ----------------
__global__ void k_f32_to_f16(const float* __restrict__ s, _Float16* __restrict__ d, int n) {
    int i = blockIdx.x * 256 + threadIdx.x;
    if (i < n) d[i] = (_Float16)s[i];
}

// pad x_proj_w (K,44,DN) -> f16 (K,48,DN) with zero rows 44..47
__global__ void k_pad_xproj(const float* __restrict__ s, _Float16* __restrict__ d) {
    int i = blockIdx.x * 256 + threadIdx.x;
    int total = K_ * NPAD * DN_;
    if (i >= total) return;
    int dd = i % DN_;
    int n  = (i / DN_) % NPAD;
    int k  = i / (DN_ * NPAD);
    float v = (n < R_ + 2 * NS_) ? s[((size_t)k * (R_ + 2 * NS_) + n) * DN_ + dd] : 0.0f;
    d[i] = (_Float16)v;
}

// ---------------- depthwise convs ----------------
// x (B,C,H,W) f32 -> u (B,H,W,C) f32
__global__ void k_dwconv1(const float* __restrict__ x, const float* __restrict__ w,
                          const float* __restrict__ bias, float* __restrict__ u) {
    int i = blockIdx.x * 256 + threadIdx.x;
    if (i >= B_ * C_ * H_ * W_) return;
    int wx = i % W_;
    int hy = (i / W_) % H_;
    int c  = (i / (W_ * H_)) % C_;
    int b  = i / (W_ * H_ * C_);
    const float* xp = x + (size_t)(b * C_ + c) * H_ * W_;
    const float* wp = w + c * 9;
    float acc = bias[c];
#pragma unroll
    for (int ky = 0; ky < 3; ++ky) {
        int yy = hy + ky - 1;
        if (yy < 0 || yy >= H_) continue;
#pragma unroll
        for (int kx = 0; kx < 3; ++kx) {
            int xx = wx + kx - 1;
            if (xx < 0 || xx >= W_) continue;
            acc += xp[yy * W_ + xx] * wp[ky * 3 + kx];
        }
    }
    u[((size_t)(b * H_ + hy) * W_ + wx) * C_ + c] = acc;
}

// xin channel of xz (M,2*DN) -> depthwise conv + SiLU -> xc (B,DN,H,W)
__global__ void k_dwconv2_silu(const float* __restrict__ xz, const float* __restrict__ w,
                               const float* __restrict__ bias, float* __restrict__ xc) {
    int i = blockIdx.x * 256 + threadIdx.x;
    if (i >= B_ * DN_ * H_ * W_) return;
    int wx = i % W_;
    int hy = (i / W_) % H_;
    int d  = (i / (W_ * H_)) % DN_;
    int b  = i / (W_ * H_ * DN_);
    const float* wp = w + d * 9;
    float acc = bias[d];
#pragma unroll
    for (int ky = 0; ky < 3; ++ky) {
        int yy = hy + ky - 1;
        if (yy < 0 || yy >= H_) continue;
#pragma unroll
        for (int kx = 0; kx < 3; ++kx) {
            int xx = wx + kx - 1;
            if (xx < 0 || xx >= W_) continue;
            acc += xz[((size_t)b * L_ + yy * W_ + xx) * (2 * DN_) + d] * wp[ky * 3 + kx];
        }
    }
    xc[i] = silu_f(acc);
}

// ---------------- LayerNorm -> f16 ----------------
__global__ void k_ln_f16(const float* __restrict__ in, const float* __restrict__ g,
                         const float* __restrict__ bt, _Float16* __restrict__ out, int Cdim) {
    int tok = blockIdx.x;
    int tid = threadIdx.x;
    __shared__ float sv[DN_];
    __shared__ float r0[256];
    __shared__ float r1[256];
    const float* p = in + (size_t)tok * Cdim;
    float s = 0.f, s2 = 0.f;
    for (int d = tid; d < Cdim; d += 256) {
        float v = p[d];
        sv[d] = v;
        s += v;
        s2 += v * v;
    }
    r0[tid] = s; r1[tid] = s2;
    __syncthreads();
    for (int st = 128; st > 0; st >>= 1) {
        if (tid < st) { r0[tid] += r0[tid + st]; r1[tid] += r1[tid + st]; }
        __syncthreads();
    }
    float mean = r0[0] / Cdim;
    float var  = r1[0] / Cdim - mean * mean;
    float rstd = rsqrtf(var + 1e-5f);
    for (int d = tid; d < Cdim; d += 256)
        out[(size_t)tok * Cdim + d] = (_Float16)(((sv[d] - mean) * rstd) * g[d] + bt[d]);
}

// ---------------- generic WMMA GEMM ----------------
// out[M,N] = A[M,K] * W[N,K]^T  (+bias, act, +resid), f16 inputs, f32 acc
// block = 256 threads (8 waves), each wave owns one 16x16 tile; block covers 128 M-rows.
// A fragment layout per dense 16-bit A table; B fragment per dense-B (sparse doc) table.
__global__ void __launch_bounds__(256)
k_gemm_wmma(const _Float16* __restrict__ A, const _Float16* __restrict__ W,
            const float* __restrict__ bias, float* __restrict__ out32,
            _Float16* __restrict__ out16, const float* __restrict__ resid,
            int M, int N, int Kd,
            long long sAb, long long sBb, long long sCb, int act) {
    int bz = blockIdx.z;
    A += (size_t)bz * sAb;
    W += (size_t)bz * sBb;
    size_t cOff = (size_t)bz * sCb;

    int wave = threadIdx.x >> 5;
    int lane = threadIdx.x & 31;
    int khalf = lane >> 4;
    int row   = lane & 15;
    int tileM = blockIdx.y * 128 + wave * 16;
    int tileN = blockIdx.x * 16;

    const _Float16* Arow = A + (size_t)(tileM + row) * Kd;
    const _Float16* Wrow = W + (size_t)(tileN + row) * Kd;

    union { v16h v; v8h h[2]; } a;
    union { v16h v; v8h h[2]; } b;
    v8f acc = {};

    for (int k0 = 0; k0 < Kd; k0 += 32) {
        // A 16x32: lanes 0-15 hold K = {k0..k0+7, k0+16..k0+23}; lanes 16-31 shifted by 8
        a.h[0] = *(const v8h*)(Arow + k0 + khalf * 8);
        a.h[1] = *(const v8h*)(Arow + k0 + 16 + khalf * 8);
        // B 32x16: same K order as A: lanes 0-15 hold K = {k0..k0+7, k0+16..k0+23}; lanes 16-31 shifted by 8
        b.h[0] = *(const v8h*)(Wrow + k0 + khalf * 8);
        b.h[1] = *(const v8h*)(Wrow + k0 + 16 + khalf * 8);
        acc = __builtin_amdgcn_wmma_f32_16x16x32_f16(false, a.v, false, b.v,
                                                     (short)0, acc, false, false);
    }

    int n = tileN + row;
    float bv = bias ? bias[n] : 0.0f;
#pragma unroll
    for (int v = 0; v < 8; ++v) {
        int m = tileM + v + 8 * khalf;
        float xv = acc[v] + bv;
        if (act == 1) xv = gelu_exact(xv);
        size_t idx = cOff + (size_t)m * N + n;
        if (resid) xv += resid[idx];
        if (out32) out32[idx] = xv;
        if (out16) out16[idx] = (_Float16)xv;
    }
}

// ---------------- cross-scan gather ----------------
__global__ void k_cross_scan(const float* __restrict__ xc, _Float16* __restrict__ xs) {
    size_t i = (size_t)blockIdx.x * 256 + threadIdx.x;
    size_t total = (size_t)B_ * K_ * L_ * DN_;
    if (i >= total) return;
    int d = (int)(i % DN_);
    int l = (int)((i / DN_) % L_);
    int k = (int)((i / ((size_t)DN_ * L_)) % K_);
    int b = (int)(i / ((size_t)DN_ * L_ * K_));
    int h, w;
    if (k == 0)      { h = l / W_;  w = l % W_; }
    else if (k == 1) { w = l / H_;  h = l % H_; }
    else if (k == 2) { int lr = L_ - 1 - l; h = lr / W_; w = lr % W_; }
    else             { int lr = L_ - 1 - l; w = lr / H_; h = lr % H_; }
    xs[i] = (_Float16)xc[((size_t)(b * DN_ + d) * H_ + h) * W_ + w];
}

// ---------------- dt projection + softplus ----------------
__global__ void k_dtproj(const float* __restrict__ dbl, const float* __restrict__ wdt,
                         const float* __restrict__ bdt, float* __restrict__ dtf) {
    size_t i = (size_t)blockIdx.x * 256 + threadIdx.x;
    size_t total = (size_t)B_ * K_ * L_ * DN_;
    if (i >= total) return;
    int d = (int)(i % DN_);
    size_t bkl = i / DN_;
    int k = (int)((bkl / L_) % K_);
    const float* r  = dbl + bkl * NPAD;
    const float* wp = wdt + ((size_t)k * DN_ + d) * R_;
    float acc = bdt[k * DN_ + d];
#pragma unroll
    for (int j = 0; j < R_; ++j) acc += r[j] * wp[j];
    dtf[i] = (acc > 20.f) ? acc : log1pf(__expf(acc));
}

// ---------------- selective scan (one lane per (b,k,d)) ----------------
// dty holds dt on input and is overwritten with y in place (same element touched once).
__global__ void k_scan(const float* __restrict__ Alog, const float* __restrict__ Dsv,
                       const float* __restrict__ dbl, const _Float16* __restrict__ xs,
                       float* __restrict__ dty) {
    int t = blockIdx.x * 128 + threadIdx.x;
    if (t >= B_ * K_ * DN_) return;
    int d = t % DN_;
    int k = (t / DN_) % K_;
    int b = t / (DN_ * K_);
    float Ar[NS_], hs[NS_];
#pragma unroll
    for (int n = 0; n < NS_; ++n) {
        Ar[n] = -__expf(Alog[((size_t)k * DN_ + d) * NS_ + n]);
        hs[n] = 0.0f;
    }
    float Dd = Dsv[k * DN_ + d];
    size_t base = (size_t)(b * K_ + k) * L_;
    for (int l = 0; l < L_; ++l) {
        size_t bl = base + l;
        float dt = dty[bl * DN_ + d];
        float x  = (float)xs[bl * DN_ + d];
        const float* bc = dbl + bl * NPAD;
        float dtx = dt * x;
        float y = 0.0f;
#pragma unroll
        for (int n = 0; n < NS_; ++n) {
            hs[n] = hs[n] * __expf(dt * Ar[n]) + dtx * bc[R_ + n];
            y += hs[n] * bc[R_ + NS_ + n];
        }
        dty[bl * DN_ + d] = y + Dd * x;
    }
}

// ---------------- cross-merge + out_norm + SiLU gate -> f16 ----------------
__global__ void k_merge_gate(const float* __restrict__ ys, const float* __restrict__ xz,
                             const float* __restrict__ g, const float* __restrict__ bt,
                             _Float16* __restrict__ yg) {
    int tok = blockIdx.x;
    int tid = threadIdx.x;
    int b = tok / L_;
    int pos = tok % L_;
    int h = pos / W_, w = pos % W_;
    int tp = w * H_ + h;
    __shared__ float sv[DN_];
    __shared__ float r0[256];
    __shared__ float r1[256];
    size_t b0 = (size_t)(b * K_ + 0) * L_;
    size_t b1 = (size_t)(b * K_ + 1) * L_;
    size_t b2 = (size_t)(b * K_ + 2) * L_;
    size_t b3 = (size_t)(b * K_ + 3) * L_;
    float s = 0.f, s2 = 0.f;
    for (int d = tid; d < DN_; d += 256) {
        float y = ys[(b0 + pos) * DN_ + d]
                + ys[(b2 + (L_ - 1 - pos)) * DN_ + d]
                + ys[(b1 + tp) * DN_ + d]
                + ys[(b3 + (L_ - 1 - tp)) * DN_ + d];
        sv[d] = y;
        s += y;
        s2 += y * y;
    }
    r0[tid] = s; r1[tid] = s2;
    __syncthreads();
    for (int st = 128; st > 0; st >>= 1) {
        if (tid < st) { r0[tid] += r0[tid + st]; r1[tid] += r1[tid + st]; }
        __syncthreads();
    }
    float mean = r0[0] / DN_;
    float var  = r1[0] / DN_ - mean * mean;
    float rstd = rsqrtf(var + 1e-5f);
    for (int d = tid; d < DN_; d += 256) {
        float z = xz[(size_t)tok * (2 * DN_) + DN_ + d];
        float val = ((sv[d] - mean) * rstd * g[d] + bt[d]) * silu_f(z);
        yg[(size_t)tok * DN_ + d] = (_Float16)val;
    }
}

// ---------------- final GELU + NHWC->NCHW ----------------
__global__ void k_final_gelu(const float* __restrict__ u, float* __restrict__ out) {
    int i = blockIdx.x * 256 + threadIdx.x;
    if (i >= B_ * C_ * H_ * W_) return;
    int w = i % W_;
    int h = (i / W_) % H_;
    int c = (i / (H_ * W_)) % C_;
    int b = i / (C_ * H_ * W_);
    float v = u[((size_t)(b * H_ + h) * W_ + w) * C_ + c];
    out[i] = gelu_exact(v);
}

extern "C" void kernel_launch(void* const* d_in, const int* in_sizes, int n_in,
                              void* d_out, int out_size, void* d_ws, size_t ws_size,
                              hipStream_t stream) {
    (void)in_sizes; (void)n_in; (void)out_size; (void)ws_size;
    const float* x         = (const float*)d_in[0];
    const float* conv_w    = (const float*)d_in[1];
    const float* conv_b    = (const float*)d_in[2];
    const float* ln1_g     = (const float*)d_in[3];
    const float* ln1_b     = (const float*)d_in[4];
    const float* in_proj_w = (const float*)d_in[5];
    const float* conv2_w   = (const float*)d_in[6];
    const float* conv2_b   = (const float*)d_in[7];
    const float* x_proj_w  = (const float*)d_in[8];
    const float* dt_proj_w = (const float*)d_in[9];
    const float* dt_proj_b = (const float*)d_in[10];
    const float* A_log     = (const float*)d_in[11];
    const float* Ds        = (const float*)d_in[12];
    const float* onorm_g   = (const float*)d_in[13];
    const float* onorm_b   = (const float*)d_in[14];
    const float* out_proj_w= (const float*)d_in[15];
    const float* ln2_g     = (const float*)d_in[16];
    const float* ln2_b     = (const float*)d_in[17];
    const float* fc1_w     = (const float*)d_in[18];
    const float* fc1_b     = (const float*)d_in[19];
    const float* fc2_w     = (const float*)d_in[20];
    const float* fc2_b     = (const float*)d_in[21];
    float* out = (float*)d_out;

    // workspace layout
    char* ws = (char*)d_ws;
    size_t off = 0;
    auto alloc = [&](size_t bytes) -> char* {
        size_t o = (off + 255) & ~(size_t)255;
        off = o + bytes;
        return ws + o;
    };
    float*    u     = (float*)alloc((size_t)MTOK * C_ * 4);
    _Float16* h16   = (_Float16*)alloc((size_t)MTOK * C_ * 2);       // ln1 out, reused for ln2 out
    float*    xz    = (float*)alloc((size_t)MTOK * 2 * DN_ * 4);
    float*    xc    = (float*)alloc((size_t)B_ * DN_ * H_ * W_ * 4);
    _Float16* xs16  = (_Float16*)alloc((size_t)B_ * K_ * L_ * DN_ * 2);
    float*    dbl   = (float*)alloc((size_t)B_ * K_ * L_ * NPAD * 4);
    float*    dtf   = (float*)alloc((size_t)B_ * K_ * L_ * DN_ * 4); // dt in, y (ys) out
    _Float16* w_in  = (_Float16*)alloc((size_t)2 * DN_ * C_ * 2);
    _Float16* w_x   = (_Float16*)alloc((size_t)K_ * NPAD * DN_ * 2);
    _Float16* w_out = (_Float16*)alloc((size_t)C_ * DN_ * 2);
    _Float16* w_fc1 = (_Float16*)alloc((size_t)MH_ * C_ * 2);
    _Float16* w_fc2 = (_Float16*)alloc((size_t)C_ * MH_ * 2);
    _Float16* g116  = xs16;            // alias: xs16 dead after scan
    _Float16* yg16  = (_Float16*)dbl;  // alias: dbl dead after scan

    dim3 blk(256);
    auto g1 = [](int n) { return dim3((n + 255) / 256); };

    // weight conversions
    k_f32_to_f16<<<g1(2 * DN_ * C_), blk, 0, stream>>>(in_proj_w, w_in, 2 * DN_ * C_);
    k_pad_xproj<<<g1(K_ * NPAD * DN_), blk, 0, stream>>>(x_proj_w, w_x);
    k_f32_to_f16<<<g1(C_ * DN_), blk, 0, stream>>>(out_proj_w, w_out, C_ * DN_);
    k_f32_to_f16<<<g1(MH_ * C_), blk, 0, stream>>>(fc1_w, w_fc1, MH_ * C_);
    k_f32_to_f16<<<g1(C_ * MH_), blk, 0, stream>>>(fc2_w, w_fc2, C_ * MH_);

    // dwconv1 -> u (NHWC)
    k_dwconv1<<<g1(B_ * C_ * H_ * W_), blk, 0, stream>>>(x, conv_w, conv_b, u);
    // LN1 -> h16
    k_ln_f16<<<dim3(MTOK), blk, 0, stream>>>(u, ln1_g, ln1_b, h16, C_);
    // in_proj: xz = h16 @ w_in^T
    k_gemm_wmma<<<dim3(2 * DN_ / 16, MTOK / 128, 1), blk, 0, stream>>>(
        h16, w_in, nullptr, xz, nullptr, nullptr, MTOK, 2 * DN_, C_, 0, 0, 0, 0);
    // dwconv2 + silu -> xc (NCHW)
    k_dwconv2_silu<<<g1(B_ * DN_ * H_ * W_), blk, 0, stream>>>(xz, conv2_w, conv2_b, xc);
    // cross-scan -> xs16 (B,K,L,DN) f16
    k_cross_scan<<<g1(B_ * K_ * L_ * DN_), blk, 0, stream>>>(xc, xs16);
    // x_proj batched GEMM: dbl = xs16 @ w_x^T   (batch = B*K)
    k_gemm_wmma<<<dim3(NPAD / 16, L_ / 128, B_ * K_), blk, 0, stream>>>(
        xs16, w_x, nullptr, dbl, nullptr, nullptr, L_, NPAD, DN_,
        (long long)L_ * DN_, (long long)NPAD * DN_, (long long)L_ * NPAD, 0);
    // dt_proj + softplus -> dtf
    k_dtproj<<<g1(B_ * K_ * L_ * DN_), blk, 0, stream>>>(dbl, dt_proj_w, dt_proj_b, dtf);
    // selective scan (dtf -> ys in place)
    k_scan<<<dim3((B_ * K_ * DN_ + 127) / 128), dim3(128), 0, stream>>>(A_log, Ds, dbl, xs16, dtf);
    // cross-merge + out_norm + silu(z) gate -> yg16
    k_merge_gate<<<dim3(MTOK), blk, 0, stream>>>(dtf, xz, onorm_g, onorm_b, yg16);
    // out_proj with residual add into u
    k_gemm_wmma<<<dim3(C_ / 16, MTOK / 128, 1), blk, 0, stream>>>(
        yg16, w_out, nullptr, u, nullptr, u, MTOK, C_, DN_, 0, 0, 0, 0);
    // LN2 -> h16
    k_ln_f16<<<dim3(MTOK), blk, 0, stream>>>(u, ln2_g, ln2_b, h16, C_);
    // fc1 + bias + gelu -> f16
    k_gemm_wmma<<<dim3(MH_ / 16, MTOK / 128, 1), blk, 0, stream>>>(
        h16, w_fc1, fc1_b, nullptr, g116, nullptr, MTOK, MH_, C_, 0, 0, 0, 1);
    // fc2 + bias + residual into u
    k_gemm_wmma<<<dim3(C_ / 16, MTOK / 128, 1), blk, 0, stream>>>(
        g116, w_fc2, fc2_b, u, nullptr, u, MTOK, C_, MH_, 0, 0, 0, 0);
    // final gelu + NHWC->NCHW
    k_final_gelu<<<g1(B_ * C_ * H_ * W_), blk, 0, stream>>>(u, out);
}
